// SwinTransformerBlock_1589137899678
// MI455X (gfx1250) — matrix operand order repaired and checked by reading, not compiled, not run on hardware
//
#include <hip/hip_runtime.h>
#include <hip/hip_bf16.h>
#include <math.h>

typedef __attribute__((ext_vector_type(16))) _Float16 v16h;
typedef __attribute__((ext_vector_type(8)))  _Float16 v8h;
typedef __attribute__((ext_vector_type(8)))  float    v8f;
typedef __attribute__((ext_vector_type(4)))  int      v4i;

union AFrag { v16h v; v8h h[2]; };

__device__ inline v8f zero_v8f() {
    v8f z;
#pragma unroll
    for (int i = 0; i < 8; ++i) z[i] = 0.0f;
    return z;
}

#define GLOBAL_AS __attribute__((address_space(1)))
#define LDS_AS    __attribute__((address_space(3)))

#if __has_builtin(__builtin_amdgcn_global_load_async_to_lds_b128)
#define HAVE_ASYNC_LDS 1
#else
#define HAVE_ASYNC_LDS 0
#endif

__device__ inline void wait_asynccnt0() {
#if HAVE_ASYNC_LDS
#if __has_builtin(__builtin_amdgcn_s_wait_asynccnt)
    __builtin_amdgcn_s_wait_asynccnt(0);
#else
    asm volatile("s_wait_asynccnt 0" ::: "memory");
#endif
#endif
}

#define TOKENS 100352      // 32*56*56 = 2048 windows * 49
#define CDIM   384
#define NWIN   2048
#define NHEAD  12
#define HDIM   32
#define NSEQ   49
static const size_t SEG = (size_t)NWIN * NHEAD * NSEQ * HDIM;  // elems per q/k/v buffer

// ---------------------------------------------------------------------------
// Weight convert fp32 [K][N] -> f16 transposed [N][K]
// ---------------------------------------------------------------------------
__global__ void wprep(const float* __restrict__ in, _Float16* __restrict__ out,
                      int K, int N) {
    int idx = blockIdx.x * 256 + threadIdx.x;
    if (idx < K * N) {
        int k = idx / N;
        int n = idx - k * N;
        out[(size_t)n * K + k] = (_Float16)in[idx];
    }
}

// ---------------------------------------------------------------------------
// LayerNorm over C=384; one wave per token, 8 tokens per block.
// ---------------------------------------------------------------------------
template <bool WINDOWED>
__global__ void __launch_bounds__(256)
ln_kernel(const float* __restrict__ x, const float* __restrict__ g,
          const float* __restrict__ bb, _Float16* __restrict__ out) {
    int wv = threadIdx.x >> 5, lane = threadIdx.x & 31;
    size_t t = (size_t)blockIdx.x * 8 + wv;
    const float* xp = x + t * CDIM;
    float v[12];
    float s = 0.0f;
#pragma unroll
    for (int i = 0; i < 12; ++i) { v[i] = xp[lane + i * 32]; s += v[i]; }
#pragma unroll
    for (int o = 16; o > 0; o >>= 1) s += __shfl_xor(s, o, 32);
    float mean = s * (1.0f / 384.0f);
    float s2 = 0.0f;
#pragma unroll
    for (int i = 0; i < 12; ++i) { float d = v[i] - mean; s2 += d * d; }
#pragma unroll
    for (int o = 16; o > 0; o >>= 1) s2 += __shfl_xor(s2, o, 32);
    float rstd = rsqrtf(s2 * (1.0f / 384.0f) + 1e-5f);

    size_t orow;
    if (WINDOWED) {
        int tt = (int)t;
        int b = tt / 3136;
        int rem = tt - b * 3136;
        int h = rem / 56;
        int w = rem - h * 56;
        int wi = b * 64 + (h / 7) * 8 + (w / 7);
        int n = (h % 7) * 7 + (w % 7);
        orow = (size_t)wi * NSEQ + n;
    } else {
        orow = t;
    }
    _Float16* op = out + orow * CDIM;
#pragma unroll
    for (int i = 0; i < 12; ++i) {
        int c = lane + i * 32;
        op[c] = (_Float16)((v[i] - mean) * rstd * g[c] + bb[c]);
    }
}

// ---------------------------------------------------------------------------
// Generic WMMA GEMM: C[M x NTOT] = A[M x K] @ Wt^T + bias, 64x64 tile/block.
// K, NTOT are compile-time; the pipelined mainloop is unrolled by 2.
// B tile (64 cols x 32 K, 4 KB) double-buffered in LDS via
// GLOBAL_LOAD_ASYNC_TO_LDS_B128 (ASYNCcnt); A fragments prefetched one
// k-step ahead in registers.
// EPI 0: qkv scatter (f16, scale q)   EPI 1: proj + residual + unwindow (f32)
// EPI 2: exact GELU (f16)             EPI 3: accumulate into f32 out
// ---------------------------------------------------------------------------
template <int EPI, int K, int NTOT>
__global__ void __launch_bounds__(128)
gemm64(const _Float16* __restrict__ A, const _Float16* __restrict__ Wt,
       const float* __restrict__ bias, _Float16* __restrict__ outH,
       float* __restrict__ outF, const float* __restrict__ xres) {
    __shared__ __align__(64) _Float16 BT[2][64][32];

    constexpr int tiles_n = NTOT >> 6;
    const int tm = blockIdx.x / tiles_n;
    const int tn = blockIdx.x % tiles_n;
    const int tid = threadIdx.x;
    const int wv = tid >> 5;
    const int lane = tid & 31;
    const int l15 = lane & 15;
    const int hi = (lane >> 4) & 1;
    const int koff_a = hi * 16;
    const int koff_b = hi * 16;

    const size_t arow = (size_t)tm * 64 + wv * 16 + l15;
    const _Float16* ap = A + arow * K;

    // cooperative B-tile fill: thread t -> column t/2, half (t&1)*16
    const int bcol = tid >> 1;
    const int bhalf = (tid & 1) * 16;
    const _Float16* bsrc0 = Wt + (size_t)(tn * 64 + bcol) * K + bhalf;

    auto issueB = [&](int buf, int k0) {
        const _Float16* gsrc = bsrc0 + k0;
        _Float16* ldst = &BT[buf][bcol][bhalf];
#if HAVE_ASYNC_LDS
        __builtin_amdgcn_global_load_async_to_lds_b128(
            (GLOBAL_AS v4i*)gsrc, (LDS_AS v4i*)ldst, 0, 0);
        __builtin_amdgcn_global_load_async_to_lds_b128(
            (GLOBAL_AS v4i*)(gsrc + 8), (LDS_AS v4i*)(ldst + 8), 0, 0);
#else
        *(v8h*)ldst = *(const v8h*)gsrc;
        *(v8h*)(ldst + 8) = *(const v8h*)(gsrc + 8);
#endif
    };
    auto loadA = [&](int k0) {
        AFrag a;
        a.h[0] = *(const v8h*)(ap + k0 + koff_a);
        a.h[1] = *(const v8h*)(ap + k0 + 8 + koff_a);
        return a;
    };

    v8f acc[4];
#pragma unroll
    for (int j = 0; j < 4; ++j) acc[j] = zero_v8f();

    issueB(0, 0);
    AFrag a = loadA(0);
    wait_asynccnt0();
    __syncthreads();

    constexpr int nsteps = K >> 5;
#pragma unroll 2
    for (int i = 0; i < nsteps; ++i) {
        const int buf = i & 1;
        const bool more = (i + 1) < nsteps;
        AFrag an;
        if (more) {
            issueB(buf ^ 1, (i + 1) << 5);
            an = loadA((i + 1) << 5);
            __builtin_prefetch(ap + ((i + 2) << 5), 0, 0);
        }
#pragma unroll
        for (int j = 0; j < 4; ++j) {
            v16h b = *(const v16h*)(&BT[buf][j * 16 + l15][koff_b]);
            acc[j] = __builtin_amdgcn_wmma_f32_16x16x32_f16(
                false, a.v, false, b, (short)0, acc[j], false, false);
        }
        if (more) {
            wait_asynccnt0();
            __syncthreads();
            a = an;
        }
    }

#pragma unroll
    for (int j = 0; j < 4; ++j) {
#pragma unroll
        for (int r = 0; r < 8; ++r) {
            int grow = tm * 64 + wv * 16 + r + hi * 8;
            int c = tn * 64 + j * 16 + l15;
            float val = acc[j][r] + bias[c];
            if (EPI == 0) {  // qkv scatter to [s][win*12+head][n][d]
                int s = c / CDIM;
                int rem = c - s * CDIM;
                int head = rem >> 5;
                int d = rem & 31;
                if (s == 0) val *= 0.17677669529663687f;  // 1/sqrt(32)
                int wi = grow / NSEQ;
                int n = grow - wi * NSEQ;
                outH[(size_t)s * SEG +
                     (((size_t)(wi * NHEAD + head)) * NSEQ + n) * HDIM + d] =
                    (_Float16)val;
            } else if (EPI == 1) {  // proj + residual, un-window to (b,h,w,c)
                int wi = grow / NSEQ;
                int n = grow - wi * NSEQ;
                int b = wi >> 6;
                int wrem = wi & 63;
                int wh = wrem >> 3, ww = wrem & 7;
                int r7 = n / 7, c7 = n - r7 * 7;
                int hh = wh * 7 + r7, wwp = ww * 7 + c7;
                size_t idx = (((size_t)b * 56 + hh) * 56 + wwp) * CDIM + c;
                outF[idx] = xres[idx] + val;
            } else if (EPI == 2) {  // exact GELU -> f16
                float gl = 0.5f * val * (1.0f + erff(val * 0.70710678118654752f));
                outH[(size_t)grow * NTOT + c] = (_Float16)gl;
            } else {  // EPI == 3: accumulate into f32 output
                size_t idx = (size_t)grow * CDIM + c;
                outF[idx] += val;
            }
        }
    }
}

// ---------------------------------------------------------------------------
// Attention: one block per (window, head). 128 threads = 4 waves.
// Phase 0: gather 49x49 rel-pos bias tile + transposed V into LDS.
// Phase 1: S = Q @ K^T padded to 64x64 (16 WMMAs), raw scores -> LDS.
// Phase 2: row softmax (adds bias + masks while reading S) -> f16 P.
// Phase 3: O = P @ V (4 WMMAs).
// ---------------------------------------------------------------------------
__global__ void __launch_bounds__(128)
attn_kernel(const _Float16* __restrict__ Qb, const _Float16* __restrict__ Kb,
            const _Float16* __restrict__ Vb, const float* __restrict__ btab,
            const int* __restrict__ relidx, _Float16* __restrict__ out) {
    __shared__ __align__(32) float S[64][64];
    __shared__ __align__(32) _Float16 P[64][64];
    __shared__ __align__(32) _Float16 Vt[32][64];
    __shared__ float Btile[NSEQ * NSEQ];

    int wid = blockIdx.x / NHEAD;
    int head = blockIdx.x % NHEAD;
    size_t base = ((size_t)wid * NHEAD + head) * NSEQ * HDIM;
    const _Float16* Qp = Qb + base;
    const _Float16* Kp = Kb + base;
    const _Float16* Vp = Vb + base;

    int tid = threadIdx.x;
    // bias tile gather (parallel, once per block)
    for (int idx = tid; idx < NSEQ * NSEQ; idx += 128)
        Btile[idx] = btab[relidx[idx] * NHEAD + head];
    // transposed V (zero-padded keys)
    for (int idx = tid; idx < 32 * 64; idx += 128) {
        int d = idx >> 6;
        int j = idx & 63;
        Vt[d][j] = (j < NSEQ) ? Vp[(size_t)j * HDIM + d] : (_Float16)0.0f;
    }
    __syncthreads();

    int wv = tid >> 5, lane = tid & 31, l15 = lane & 15;
    int hi = (lane >> 4) & 1;
    int koff_a = hi * 16, koff_b = hi * 16;

    // ---- phase 1: S = Q @ K^T, store raw scores (no mask/bias here) ----
    {
        int arow = wv * 16 + l15;
        int qr = arow < NSEQ ? arow : NSEQ - 1;  // clamp padded rows
        AFrag a;
        a.h[0] = *(const v8h*)(Qp + qr * HDIM + koff_a);
        a.h[1] = *(const v8h*)(Qp + qr * HDIM + 8 + koff_a);
#pragma unroll
        for (int jt = 0; jt < 4; ++jt) {
            int key = jt * 16 + l15;
            int kr = key < NSEQ ? key : NSEQ - 1;
            v16h b = *(const v16h*)(Kp + kr * HDIM + koff_b);
            v8f s = __builtin_amdgcn_wmma_f32_16x16x32_f16(
                false, a.v, false, b, (short)0, zero_v8f(), false, false);
#pragma unroll
            for (int r = 0; r < 8; ++r) {
                int srow = wv * 16 + r + hi * 8;
                int scol = jt * 16 + l15;
                S[srow][scol] = s[r];
            }
        }
    }
    __syncthreads();

    // ---- phase 2: row softmax with bias (cols >= 49 never read) ----
    if (tid < 64) {
        int row = tid;
        if (row < NSEQ) {
            float m = -1e30f;
            for (int c = 0; c < NSEQ; ++c) {
                float v = S[row][c] + Btile[row * NSEQ + c];
                S[row][c] = v;
                m = fmaxf(m, v);
            }
            float sum = 0.0f;
            for (int c = 0; c < NSEQ; ++c) {
                float e = __expf(S[row][c] - m);
                S[row][c] = e;
                sum += e;
            }
            float inv = 1.0f / sum;
            for (int c = 0; c < 64; ++c)
                P[row][c] = (c < NSEQ) ? (_Float16)(S[row][c] * inv) : (_Float16)0.0f;
        } else {
            for (int c = 0; c < 64; ++c) P[row][c] = (_Float16)0.0f;
        }
    }
    __syncthreads();

    // ---- phase 3: O = P @ V ----
    v8f acc2[2];
    acc2[0] = zero_v8f();
    acc2[1] = zero_v8f();
#pragma unroll
    for (int k0 = 0; k0 < 64; k0 += 32) {
        int prow = wv * 16 + l15;
        AFrag a;
        a.h[0] = *(const v8h*)(&P[prow][k0 + koff_a]);
        a.h[1] = *(const v8h*)(&P[prow][k0 + 8 + koff_a]);
#pragma unroll
        for (int jt = 0; jt < 2; ++jt) {
            int d = jt * 16 + l15;
            v16h b = *(const v16h*)(&Vt[d][k0 + koff_b]);
            acc2[jt] = __builtin_amdgcn_wmma_f32_16x16x32_f16(
                false, a.v, false, b, (short)0, acc2[jt], false, false);
        }
    }
#pragma unroll
    for (int jt = 0; jt < 2; ++jt) {
#pragma unroll
        for (int r = 0; r < 8; ++r) {
            int row = wv * 16 + r + hi * 8;
            if (row < NSEQ) {
                int d = jt * 16 + l15;
                out[((size_t)wid * NSEQ + row) * CDIM + head * HDIM + d] =
                    (_Float16)acc2[jt][r];
            }
        }
    }
}

// ---------------------------------------------------------------------------
extern "C" void kernel_launch(void* const* d_in, const int* in_sizes, int n_in,
                              void* d_out, int out_size, void* d_ws,
                              size_t ws_size, hipStream_t stream) {
    (void)in_sizes; (void)n_in; (void)out_size; (void)ws_size;
    const float* x      = (const float*)d_in[0];
    const float* ln1_g  = (const float*)d_in[1];
    const float* ln1_b  = (const float*)d_in[2];
    const float* qkv_w  = (const float*)d_in[3];
    const float* qkv_b  = (const float*)d_in[4];
    const float* btab   = (const float*)d_in[5];
    const float* proj_w = (const float*)d_in[6];
    const float* proj_b = (const float*)d_in[7];
    const float* ln2_g  = (const float*)d_in[8];
    const float* ln2_b  = (const float*)d_in[9];
    const float* mlp_w1 = (const float*)d_in[10];
    const float* mlp_b1 = (const float*)d_in[11];
    const float* mlp_w2 = (const float*)d_in[12];
    const float* mlp_b2 = (const float*)d_in[13];
    const int*   relidx = (const int*)d_in[14];
    float* outF = (float*)d_out;

    char* ws = (char*)d_ws;
    size_t off = 0;
    auto alloc = [&](size_t bytes) -> void* {
        off = (off + 255) & ~(size_t)255;
        void* p = ws + off;
        off += bytes;
        return p;
    };
    _Float16* qkv_wt = (_Float16*)alloc((size_t)1152 * 384 * 2);
    _Float16* proj_wt = (_Float16*)alloc((size_t)384 * 384 * 2);
    _Float16* w1t = (_Float16*)alloc((size_t)384 * 1536 * 2);   // [1536][384]
    _Float16* w2t = (_Float16*)alloc((size_t)1536 * 384 * 2);   // [384][1536]
    _Float16* xln = (_Float16*)alloc((size_t)TOKENS * CDIM * 2);
    _Float16* qkvb = (_Float16*)alloc(3 * SEG * 2 + 4096);
    _Float16* attn = (_Float16*)alloc((size_t)TOKENS * CDIM * 2);
    _Float16* hbuf = (_Float16*)alloc((size_t)TOKENS * 1536 * 2);

    // weight prep
    wprep<<<(384 * 1152 + 255) / 256, 256, 0, stream>>>(qkv_w, qkv_wt, 384, 1152);
    wprep<<<(384 * 384 + 255) / 256, 256, 0, stream>>>(proj_w, proj_wt, 384, 384);
    wprep<<<(384 * 1536 + 255) / 256, 256, 0, stream>>>(mlp_w1, w1t, 384, 1536);
    wprep<<<(1536 * 384 + 255) / 256, 256, 0, stream>>>(mlp_w2, w2t, 1536, 384);

    // LN1 + window partition -> f16
    ln_kernel<true><<<TOKENS / 8, 256, 0, stream>>>(x, ln1_g, ln1_b, xln);
    // QKV GEMM (M=100352, K=384, N=1152)
    gemm64<0, 384, 1152><<<1568 * 18, 128, 0, stream>>>(xln, qkv_wt, qkv_b,
                                                        qkvb, nullptr, nullptr);
    // attention per (window, head)
    attn_kernel<<<NWIN * NHEAD, 128, 0, stream>>>(qkvb, qkvb + SEG,
                                                  qkvb + 2 * SEG, btab, relidx,
                                                  attn);
    // proj GEMM + residual + un-window (writes all of d_out)
    gemm64<1, 384, 384><<<1568 * 6, 128, 0, stream>>>(attn, proj_wt, proj_b,
                                                      nullptr, outF, x);
    // LN2 -> f16 (reuse xln)
    ln_kernel<false><<<TOKENS / 8, 256, 0, stream>>>(outF, ln2_g, ln2_b, xln);
    // MLP1 + GELU (M=100352, K=384, N=1536)
    gemm64<2, 384, 1536><<<1568 * 24, 128, 0, stream>>>(xln, w1t, mlp_b1, hbuf,
                                                        nullptr, nullptr);
    // MLP2 accumulate into d_out (M=100352, K=1536, N=384)
    gemm64<3, 1536, 384><<<1568 * 6, 128, 0, stream>>>(hbuf, w2t, mlp_b2,
                                                       nullptr, outF, nullptr);
}
